// S4MaskNet_90941637525629
// MI455X (gfx1250) — matrix operand order repaired and checked by reading, not compiled, not run on hardware
//
#include <hip/hip_runtime.h>
#include <hip/hip_bf16.h>
#include <cmath>

// ---------------------------------------------------------------------------
// S4MaskNet forward for MI455X (gfx1250, wave32).
//  - SSM conv as exact complex recurrence (no FFT): 1 wave per (b,h), 1 lane
//    per mode (N=32 == wave32), shfl_xor butterfly reduction per step.
//  - Dense layers via v_wmma_f32_16x16x32_bf16, fed from LDS panels staged
//    with global_load_async_to_lds_b128 (ASYNCcnt) — whole K=256 resident.
// ---------------------------------------------------------------------------

#define D_MODEL 256
#define N_STATE 32
#define BATCH_  8
#define SEQ_L   2048
#define M_TOK   (BATCH_ * SEQ_L)   // 16384 tokens
#define SPK_    2
#define GK      256                // K extent of every GEMM (== D_MODEL)

typedef __attribute__((ext_vector_type(16))) __bf16 v16bf;
typedef __attribute__((ext_vector_type(8)))  float  v8f;

__device__ __forceinline__ float gelu_exact(float x) {
    return 0.5f * x * (1.0f + erff(x * 0.70710678118654752440f));
}

// -------- CDNA5 async global->LDS copy (16B per lane), ASYNCcnt-tracked ----
__device__ __forceinline__ void async_copy_b128(const float* gsrc, const float* ldst) {
    unsigned loff = (unsigned)(unsigned long long)ldst;   // LDS byte offset
    asm volatile("global_load_async_to_lds_b128 %0, %1, off"
                 :: "v"(loff), "v"(gsrc) : "memory");
}
__device__ __forceinline__ void wait_async0() {
    asm volatile("s_wait_asynccnt 0" ::: "memory");
}

// ---------------- transpose x[b,d,l] -> u[(b*L+l), d] ----------------------
__global__ void k_transpose_in(const float* __restrict__ x, float* __restrict__ u) {
    int idx = blockIdx.x * blockDim.x + threadIdx.x;
    if (idx >= M_TOK * D_MODEL) return;
    int d = idx % D_MODEL;
    int t = idx / D_MODEL;
    int b = t / SEQ_L, l = t % SEQ_L;
    u[idx] = x[((size_t)b * D_MODEL + d) * SEQ_L + l];
}

// ---------------- layernorm over D=256, one token per block ----------------
__global__ void k_layernorm(const float* __restrict__ in, float* __restrict__ out,
                            const float* __restrict__ g, const float* __restrict__ b) {
    __shared__ float red[D_MODEL];
    int t = blockIdx.x, d = threadIdx.x;
    float v = in[(size_t)t * D_MODEL + d];
    red[d] = v; __syncthreads();
    for (int s = 128; s > 0; s >>= 1) { if (d < s) red[d] += red[d + s]; __syncthreads(); }
    float mean = red[0] * (1.0f / D_MODEL);
    __syncthreads();
    float c = v - mean;
    red[d] = c * c; __syncthreads();
    for (int s = 128; s > 0; s >>= 1) { if (d < s) red[d] += red[d + s]; __syncthreads(); }
    float var = red[0] * (1.0f / D_MODEL);
    out[(size_t)t * D_MODEL + d] = c * rsqrtf(var + 1e-5f) * g[d] + b[d];
}

// ---------------- SSM parameter precompute (per layer) ---------------------
// P layout: [4][D*N] = wre, wim, 2*CBre, 2*CBim
__global__ void k_ssm_setup(const float* __restrict__ log_dt, const float* __restrict__ Alr,
                            const float* __restrict__ Aim,   const float* __restrict__ Bre,
                            const float* __restrict__ Bim,   const float* __restrict__ Cre,
                            const float* __restrict__ Cim,   float* __restrict__ P) {
    int idx = blockIdx.x * blockDim.x + threadIdx.x;
    const int DN = D_MODEL * N_STATE;
    if (idx >= DN) return;
    int h = idx / N_STATE;
    float dt  = expf(log_dt[h]);
    float are = -expf(Alr[idx]);
    float aim = Aim[idx];
    float mag = expf(dt * are);
    float wre = mag * cosf(dt * aim);
    float wim = mag * sinf(dt * aim);
    float ere = wre - 1.0f, eim = wim;          // exp(dtA)-1
    float den = are * are + aim * aim;
    float fre = (ere * are + eim * aim) / den;  // (exp(dtA)-1)/A
    float fim = (eim * are - ere * aim) / den;
    float br = Bre[idx], bi = Bim[idx];
    float dbre = br * fre - bi * fim;
    float dbim = br * fim + bi * fre;
    float cr = Cre[idx], ci = Cim[idx];
    float cbre = cr * dbre - ci * dbim;
    float cbim = cr * dbim + ci * dbre;
    P[0 * DN + idx] = wre;
    P[1 * DN + idx] = wim;
    P[2 * DN + idx] = 2.0f * cbre;
    P[3 * DN + idx] = 2.0f * cbim;
}

// ---------------- SSM scan: 1 wave per (b,h), 1 lane per mode --------------
__global__ void k_ssm_scan(const float* __restrict__ U, const float* __restrict__ P,
                           const float* __restrict__ Dp, float* __restrict__ Y) {
    int wave = (blockIdx.x * blockDim.x + threadIdx.x) >> 5;
    int lane = threadIdx.x & 31;
    int b = wave / D_MODEL, h = wave % D_MODEL;
    const int DN = D_MODEL * N_STATE;
    int idx = h * N_STATE + lane;
    float wre = P[idx], wim = P[DN + idx];
    float cbr = P[2 * DN + idx], cbi = P[3 * DN + idx];
    float dph = Dp[h];
    float sre = 0.0f, sim = 0.0f;
    const float* ub = U + (size_t)b * SEQ_L * D_MODEL + h;
    float*       yb = Y + (size_t)b * SEQ_L * D_MODEL + h;
    for (int l = 0; l < SEQ_L; ++l) {
        float uv = ub[(size_t)l * D_MODEL];      // broadcast read
        float tr = wre * sre - wim * sim + uv;   // S = w*S + u
        float ti = wre * sim + wim * sre;
        sre = tr; sim = ti;
        float r = cbr * sre - cbi * sim;         // Re(CB*S) per mode
        r += __shfl_xor(r, 16, 32);
        r += __shfl_xor(r,  8, 32);
        r += __shfl_xor(r,  4, 32);
        r += __shfl_xor(r,  2, 32);
        r += __shfl_xor(r,  1, 32);
        if (lane == 0) yb[(size_t)l * D_MODEL] = gelu_exact(r + dph * uv);
    }
}

// ---------------- WMMA GEMM: Y = epi( X[M,256] @ W[256,N] + bias (+R) ) ----
// Block = 2 waves. LDS holds the FULL K extent:
//   lds_a : 32 rows x 256 f32 (async b128 staged)          = 32 KB
//   lds_w : 16 cols x 256 bf16 (transposed, ds stores)     =  8 KB
// Inner loop = 8 back-to-back v_wmma_f32_16x16x32_bf16, zero barriers.
__global__ void k_gemm_wmma(const float* __restrict__ X, const float* __restrict__ W,
                            const float* __restrict__ bias, const float* __restrict__ R,
                            float* __restrict__ Y, int M, int N, int do_gelu) {
    __shared__ float          lds_a[32 * GK];        // [row][k] f32
    __shared__ unsigned short lds_w[16 * GK];        // [n][k]  bf16 bits

    int lane = threadIdx.x;                 // 0..31
    int wv   = threadIdx.y;                 // 0..1
    int tid  = wv * 32 + lane;              // 0..63
    int n0   = blockIdx.x * 16;
    int mblk = blockIdx.y * 32;
    int m0   = mblk + wv * 16;
    int half = lane >> 4;
    int rc   = lane & 15;                   // A row / B,D column within tile

    // ---- stage A panel: 32x256 f32 = 2048 x 16B, async, 32 per thread ----
    #pragma unroll
    for (int j = 0; j < 32; ++j) {
        int linear = tid + 64 * j;          // 0..2047
        int r  = linear >> 6;               // /64 xfers per row
        int c4 = linear & 63;
        async_copy_b128(X + (size_t)(mblk + r) * GK + c4 * 4,
                        &lds_a[r * GK + c4 * 4]);
    }
    // ---- stage W panel transposed: 256x16 f32 -> bf16 [n][k] -------------
    #pragma unroll
    for (int j = 0; j < 64; ++j) {
        int linear = tid + 64 * j;          // 0..4095
        int k = linear >> 4;
        int n = linear & 15;
        union { __bf16 b; unsigned short u; } cv;
        cv.b = (__bf16)W[(size_t)k * N + n0 + n];
        lds_w[n * GK + k] = cv.u;
    }
    wait_async0();
    __syncthreads();

    union { v8f v; float e[8]; } acc;
    #pragma unroll
    for (int j = 0; j < 8; ++j) acc.e[j] = 0.0f;

    const float*          arow = &lds_a[(rc) * GK];
    const unsigned short* wcol = &lds_w[(rc) * GK];

    #pragma unroll
    for (int kk = 0; kk < GK / 32; ++kk) {
        int k0 = kk * 32;
        union { v16bf v; __bf16 e[16]; } af;
        union { v16bf v; unsigned u[8]; } bf;
        #pragma unroll
        for (int i = 0; i < 8; ++i) {
            // A 16x32 bf16 frag: vgpr i<4 -> K=2i (+8 hi half); i>=4 -> 16+2(i-4) (+8 hi)
            int ka = (i < 4 ? 2 * i : 16 + 2 * (i - 4)) + (half ? 8 : 0);
            float2 p = *(const float2*)&lds_a[(wv * 16 + rc) * GK + k0 + ka];
            af.e[2 * i]     = (__bf16)p.x;
            af.e[2 * i + 1] = (__bf16)p.y;
            // B 32x16 frag: same K order as A: vgpr i<4 -> K=2i (+8 hi half); i>=4 -> 16+2(i-4) (+8 hi)
            int kb = (i < 4 ? 2 * i : 16 + 2 * (i - 4)) + (half ? 8 : 0);
            bf.u[i] = *(const unsigned*)&wcol[k0 + kb];
        }
        acc.v = __builtin_amdgcn_wmma_f32_16x16x32_bf16(
            false, af.v, false, bf.v, (short)0, acc.v, false, false);
    }
    (void)arow;

    // D layout: vgpr j holds (m = j + 8*half, n = lane&15)
    #pragma unroll
    for (int j = 0; j < 8; ++j) {
        int m = m0 + j + (half ? 8 : 0);
        int n = n0 + rc;
        float v = acc.e[j] + bias[n];
        if (R) v += R[(size_t)m * N + n];
        if (do_gelu) v = gelu_exact(v);
        Y[(size_t)m * N + n] = v;
    }
}

// ---------------- finalize: relu + reshape to [SPK, B, D, L] ---------------
__global__ void k_finalize(const float* __restrict__ O, float* __restrict__ out) {
    int idx = blockIdx.x * blockDim.x + threadIdx.x;
    if (idx >= SPK_ * BATCH_ * D_MODEL * SEQ_L) return;
    int l = idx % SEQ_L;
    int t = idx / SEQ_L;
    int d = t % D_MODEL;  t /= D_MODEL;
    int b = t % BATCH_;
    int s = t / BATCH_;
    float v = O[((size_t)b * SEQ_L + l) * (SPK_ * D_MODEL) + s * D_MODEL + d];
    out[idx] = v > 0.0f ? v : 0.0f;
}

// ---------------------------------------------------------------------------
extern "C" void kernel_launch(void* const* d_in, const int* in_sizes, int n_in,
                              void* d_out, int out_size, void* d_ws, size_t ws_size,
                              hipStream_t stream) {
    const float* x      = (const float*)d_in[0];
    const float* ln_g   = (const float*)d_in[1];
    const float* ln_b   = (const float*)d_in[2];
    const float* log_dt = (const float*)d_in[3];
    const float* Alr    = (const float*)d_in[4];
    const float* Aim    = (const float*)d_in[5];
    const float* Bre    = (const float*)d_in[6];
    const float* Bim    = (const float*)d_in[7];
    const float* Cre    = (const float*)d_in[8];
    const float* Cim    = (const float*)d_in[9];
    const float* Dp     = (const float*)d_in[10];
    const float* s4oW   = (const float*)d_in[11];
    const float* s4ob   = (const float*)d_in[12];
    const float* l1W    = (const float*)d_in[13];
    const float* l1b    = (const float*)d_in[14];
    const float* l2W    = (const float*)d_in[15];
    const float* l2b    = (const float*)d_in[16];
    const float* headW[2][2] = { { (const float*)d_in[17], (const float*)d_in[19] },
                                 { (const float*)d_in[21], (const float*)d_in[23] } };
    const float* headB[2][2] = { { (const float*)d_in[18], (const float*)d_in[20] },
                                 { (const float*)d_in[22], (const float*)d_in[24] } };
    float* out = (float*)d_out;

    // workspace (floats): 4 x [M,256] + 2 x [M,512] + params (~128.1 MB)
    const size_t MB = (size_t)M_TOK * D_MODEL;
    float* u  = (float*)d_ws;      // current layer input  [M,256]
    float* b1 = u  + MB;           // scratch              [M,256]
    float* b2 = b1 + MB;           // scratch              [M,256]
    float* b3 = b2 + MB;           // o1 (residual)        [M,256]
    float* bh = b3 + MB;           // head proj            [M,512]
    float* bo = bh + 2 * MB;       // final layer output   [M,512]
    float* pp = bo + 2 * MB;       // ssm params [4][D*N]

    const int DN = D_MODEL * N_STATE;
    dim3 blkG(32, 2);
    dim3 grd256(D_MODEL / 16, M_TOK / 32);

    k_transpose_in<<<(M_TOK * D_MODEL + 255) / 256, 256, 0, stream>>>(x, u);

    for (int i = 0; i < 2; ++i) {
        // --- LN1 ---
        k_layernorm<<<M_TOK, D_MODEL, 0, stream>>>(u, b1,
            ln_g + (size_t)(i * 2 + 0) * D_MODEL, ln_b + (size_t)(i * 2 + 0) * D_MODEL);
        // --- SSM params + scan (+Dp skip +gelu) ---
        k_ssm_setup<<<(DN + 255) / 256, 256, 0, stream>>>(
            log_dt + (size_t)i * D_MODEL, Alr + (size_t)i * DN, Aim + (size_t)i * DN,
            Bre + (size_t)i * DN, Bim + (size_t)i * DN,
            Cre + (size_t)i * DN, Cim + (size_t)i * DN, pp);
        k_ssm_scan<<<(BATCH_ * D_MODEL) / 8, 256, 0, stream>>>(b1, pp, Dp + (size_t)i * D_MODEL, b2);
        // --- s4 output linear ---
        k_gemm_wmma<<<grd256, blkG, 0, stream>>>(b2, s4oW + (size_t)i * D_MODEL * D_MODEL,
            s4ob + (size_t)i * D_MODEL, nullptr, b1, M_TOK, D_MODEL, 0);
        // --- lin1 + residual(u) -> o1 ---
        k_gemm_wmma<<<grd256, blkG, 0, stream>>>(b1, l1W + (size_t)i * D_MODEL * D_MODEL,
            l1b + (size_t)i * D_MODEL, u, b3, M_TOK, D_MODEL, 0);
        // --- LN2 ---
        k_layernorm<<<M_TOK, D_MODEL, 0, stream>>>(b3, b1,
            ln_g + (size_t)(i * 2 + 1) * D_MODEL, ln_b + (size_t)(i * 2 + 1) * D_MODEL);
        // --- lin2 + gelu -> g ---
        k_gemm_wmma<<<grd256, blkG, 0, stream>>>(b1, l2W + (size_t)i * D_MODEL * D_MODEL,
            l2b + (size_t)i * D_MODEL, nullptr, b2, M_TOK, D_MODEL, 1);
        // --- head: o = g@oW + ob + (o1@pW + pb) ---
        int Cout = (i == 0) ? D_MODEL : SPK_ * D_MODEL;
        dim3 grdC(Cout / 16, M_TOK / 32);
        k_gemm_wmma<<<grdC, blkG, 0, stream>>>(b3, headW[i][1], headB[i][1], nullptr,
                                               bh, M_TOK, Cout, 0);
        float* dst = (i == 0) ? u : bo;
        k_gemm_wmma<<<grdC, blkG, 0, stream>>>(b2, headW[i][0], headB[i][0], bh,
                                               dst, M_TOK, Cout, 0);
    }

    k_finalize<<<(SPK_ * BATCH_ * D_MODEL * SEQ_L + 255) / 256, 256, 0, stream>>>(bo, out);
}
